// NCA_609885356883
// MI455X (gfx1250) — matrix operand order repaired and checked by reading, not compiled, not run on hardware
//
#include <hip/hip_runtime.h>

typedef __attribute__((ext_vector_type(2)))  _Float16 v2h;
typedef __attribute__((ext_vector_type(4)))  _Float16 v4h;
typedef __attribute__((ext_vector_type(8)))  _Float16 v8h;
typedef __attribute__((ext_vector_type(16))) _Float16 v16h;
typedef __attribute__((ext_vector_type(8)))  float    v8f;

// ---------------------------------------------------------------------------
// One block = one 16x16 spatial tile of one batch image; 256 threads = 8 waves.
// WMMA roles: A = weights (feature-major), B = activations (pixel columns):
//   hiddenT = w1 x yT   (D rows = features, D cols = pixels)
//   outT    = w2 x hidT
// So B-fragments are single contiguous v16h LDS loads, FC1 D packs feature
// pairs into b32 stores, and FC2 D stores are coalesced along w.
// ---------------------------------------------------------------------------
#define C_IN   12
#define KDIM1  48          // 4*C_IN
#define FC     96
#define HDIM   256
#define WDIM   256
#define YSTR   80          // y row stride in halves (>=64 used, 32B-aligned, anti-conflict)
#define HSTR   112         // hid row stride in halves (>=96 used, 32B-aligned, anti-conflict)

// LDS byte offsets (all 16B aligned)
#define OFF_XS    0                      // 12*18*18*4 = 15552 (pad 15616)
#define OFF_MASK  15616                  // 256*4      = 1024
#define OFF_BIAS  16640                  // 96*4       = 384 (pad 512)
#define OFF_Y     17152                  // 256*80*2   = 40960
#define OFF_W1A   58112                  // 12*32*16*2 = 12288
#define OFF_W2A   70400                  // 3*32*16*2  = 3072
#define OFF_HID   73472                  // 256*112*2  = 57344
#define SMEM_BYTES 130816

__global__ __launch_bounds__(256)
void nca_step_kernel(const float* __restrict__ x,
                     const float* __restrict__ w1,
                     const float* __restrict__ b1,
                     const float* __restrict__ w2,
                     const float* __restrict__ rmask,
                     float* __restrict__ out) {
    extern __shared__ char smem[];
    float*    xs     = (float*)(smem + OFF_XS);     // [12][18][18] halo tile
    float*    mask_s = (float*)(smem + OFF_MASK);   // [256]
    float*    bias_s = (float*)(smem + OFF_BIAS);   // [96]
    _Float16* y_s    = (_Float16*)(smem + OFF_Y);   // [256][YSTR]
    _Float16* w1a    = (_Float16*)(smem + OFF_W1A); // 12 A-frags [32][16]
    _Float16* w2a    = (_Float16*)(smem + OFF_W2A); // 3  A-frags [32][16]
    _Float16* hid    = (_Float16*)(smem + OFF_HID); // [256][HSTR]

    const int tid  = threadIdx.x;
    const int lane = tid & 31;
    const int wid  = tid >> 5;
    const int lh   = lane >> 4;         // lane half (0/1)
    const int ll   = lane & 15;

    const int bi = blockIdx.x;          // 0..8191
    const int b  = bi >> 8;
    const int th = (bi & 255) >> 4;
    const int tw = bi & 15;

    // ---------------- Phase 0: cooperative staging ----------------
    // w1 -> A-fragments (16-bit A 16x32 layout), K padded 48->64.
// A order used here (contiguous k, B matches): lane l holds row M = l&15; halves j=0..15 -> K = kk+16*(l>>4)+j.
    for (int idx = tid; idx < 12 * 512; idx += 256) {
        int frag = idx >> 9, r = idx & 511;
        int lane_l = r >> 4, j = r & 15;
        int fb = frag >> 1, ks = frag & 1;
        int f = fb * 16 + (lane_l & 15);
        int k = ks * 32 + 16 * (lane_l >> 4) + j;
        float v = (k < KDIM1) ? w1[f * KDIM1 + k] : 0.f;
        w1a[idx] = (_Float16)v;
    }
    // w2 -> A-fragments, rows padded 12->16 with zeros.
    for (int idx = tid; idx < 3 * 512; idx += 256) {
        int ks = idx >> 9, r = idx & 511;
        int lane_l = r >> 4, j = r & 15;
        int f = lane_l & 15;
        int k = ks * 32 + 16 * (lane_l >> 4) + j;
        float v = (f < C_IN) ? w2[f * FC + k] : 0.f;
        w2a[idx] = (_Float16)v;
    }
    // x halo tile with circular wrap: xs[c][18][18]
    for (int idx = tid; idx < C_IN * 18 * 18; idx += 256) {
        int c = idx / 324, rem = idx % 324;
        int i = rem / 18, j = rem % 18;
        int gh = th * 16 + i - 1; gh = (gh < 0) ? gh + HDIM : (gh >= HDIM ? gh - HDIM : gh);
        int gw = tw * 16 + j - 1; gw = (gw < 0) ? gw + WDIM : (gw >= WDIM ? gw - WDIM : gw);
        xs[idx] = x[(((size_t)b * C_IN + c) * HDIM + gh) * WDIM + gw];
    }
    // mask tile + bias
    {
        int gh = th * 16 + (tid >> 4), gw = tw * 16 + (tid & 15);
        mask_s[tid] = rmask[((size_t)b * HDIM + gh) * WDIM + gw];
        if (tid < FC) bias_s[tid] = b1[tid];
    }
    __syncthreads();

    // ---------------- Phase 1: perception (VALU stencil) ----------------
    {
        const int m = tid;                 // pixel in tile
        const int i = m >> 4, j = m & 15;
        _Float16* yrow = y_s + m * YSTR;
#pragma unroll
        for (int c = 0; c < C_IN; ++c) {
            const float* p = xs + c * 324 + i * 18 + j;   // top-left of 3x3
            float tl = p[0],  tp = p[1],  tr = p[2];
            float lf = p[18], ce = p[19], rt = p[20];
            float bl = p[36], bo = p[37], br = p[38];
            float sx  = (tr - tl) + 2.f * (rt - lf) + (br - bl);
            float sy  = (bl - tl) + 2.f * (bo - tp) + (br - tr);
            float lap = (tl + tr + bl + br) + 2.f * (tp + lf + rt + bo) - 12.f * ce;
            v4h pk;
            pk[0] = (_Float16)ce;  pk[1] = (_Float16)sx;
            pk[2] = (_Float16)sy;  pk[3] = (_Float16)lap;
            *(v4h*)(yrow + c * 4) = pk;                   // ds_store_b64
        }
        v8h z = {};
        *(v8h*)(yrow + 48) = z;                           // zero-pad K 48..63
        *(v8h*)(yrow + 56) = z;
    }
    __syncthreads();

    // ---------------- Phase 2: FC1 (WMMA, A=w1, B=yT) + bias + relu -------
    // Wave owns pixel columns [32*wid, 32*wid+32) as two 16-wide N-tiles.
#pragma unroll
    for (int t = 0; t < 2; ++t) {
        const int pb = wid * 32 + t * 16;
        const int p  = pb + ll;           // this lane's pixel column
        // y B-fragments: lane holds 16 sequential K halves of column p.
        v16h yb0 = *(const v16h*)(y_s + p * YSTR +      16 * lh);
        v16h yb1 = *(const v16h*)(y_s + p * YSTR + 32 + 16 * lh);
#pragma unroll
        for (int fb = 0; fb < 6; ++fb) {
            v8f acc = {};
            v16h a0 = *(const v16h*)(w1a + ((fb * 2 + 0) * 32 + lane) * 16);
            acc = __builtin_amdgcn_wmma_f32_16x16x32_f16(false, a0, false, yb0,
                                                         (short)0, acc, false, false);
            v16h a1 = *(const v16h*)(w1a + ((fb * 2 + 1) * 32 + lane) * 16);
            acc = __builtin_amdgcn_wmma_f32_16x16x32_f16(false, a1, false, yb1,
                                                         (short)0, acc, false, false);
            const int f0 = fb * 16 + 8 * lh;      // feature base for this lane half
#pragma unroll
            for (int r = 0; r < 8; r += 2) {      // consecutive features -> pack
                float v0 = acc[r]     + bias_s[f0 + r];
                float v1 = acc[r + 1] + bias_s[f0 + r + 1];
                v0 = v0 > 0.f ? v0 : 0.f;
                v1 = v1 > 0.f ? v1 : 0.f;
                v2h pk; pk[0] = (_Float16)v0; pk[1] = (_Float16)v1;
                *(v2h*)(hid + p * HSTR + f0 + r) = pk;    // ds_store_b32
            }
        }
    }
    __syncthreads();

    // ---------------- Phase 3: FC2 (WMMA, A=w2, B=hidT) + masked residual --
#pragma unroll
    for (int t = 0; t < 2; ++t) {
        const int pb = wid * 32 + t * 16;
        const int p  = pb + ll;
        v8f acc = {};
#pragma unroll
        for (int ks = 0; ks < 3; ++ks) {
            v16h hb = *(const v16h*)(hid + p * HSTR + ks * 32 + 16 * lh);
            v16h aw = *(const v16h*)(w2a + (ks * 32 + lane) * 16);
            acc = __builtin_amdgcn_wmma_f32_16x16x32_f16(false, aw, false, hb,
                                                         (short)0, acc, false, false);
        }
        const int i  = p >> 4, j = p & 15;
        const float mk = (mask_s[p] >= 0.5f) ? 1.f : 0.f;
        const int cb = 8 * lh;
#pragma unroll
        for (int r = 0; r < 8; ++r) {
            int c = cb + r;                        // channel; lanes 0-15 share c
            if (c < C_IN) {
                float xv = xs[c * 324 + (i + 1) * 18 + (j + 1)];
                // lanes of a half-wave write 16 consecutive w -> coalesced
                out[(((size_t)b * C_IN + c) * HDIM + (th * 16 + i)) * WDIM
                    + (tw * 16 + j)] = xv + acc[r] * mk;
            }
        }
    }
}

extern "C" void kernel_launch(void* const* d_in, const int* in_sizes, int n_in,
                              void* d_out, int out_size, void* d_ws, size_t ws_size,
                              hipStream_t stream) {
    (void)in_sizes; (void)n_in; (void)d_ws; (void)ws_size; (void)out_size;
    const float* x     = (const float*)d_in[0];
    const float* w1    = (const float*)d_in[1];
    const float* b1    = (const float*)d_in[2];
    const float* w2    = (const float*)d_in[3];
    const float* rmask = (const float*)d_in[4];
    float* out = (float*)d_out;

    (void)hipFuncSetAttribute((const void*)nca_step_kernel,
                              hipFuncAttributeMaxDynamicSharedMemorySize, SMEM_BYTES);
    nca_step_kernel<<<dim3(32 * 16 * 16), dim3(256), SMEM_BYTES, stream>>>(
        x, w1, b1, w2, rmask, out);
}
